// VQEmbeddingEMA_15307263443359
// MI455X (gfx1250) — matrix operand order repaired and checked by reading, not compiled, not run on hardware
//
#include <hip/hip_runtime.h>
#include <hip/hip_bf16.h>

#define DECAY_F   0.999f
#define ONEM_F    0.001f
#define EPS_F     1e-5f
#define COMMIT_F  0.25f
#define ORTHO_WF  10.0f

#define MC 2048      // codebook size
#define DD 512       // dim
#define NN 32768     // B*T
#define ND_TOT 16777216  // NN*DD

typedef __bf16 bf16_t;
typedef __attribute__((ext_vector_type(16))) __bf16 v16bf;
typedef __attribute__((ext_vector_type(8)))  __bf16 v8bf;
typedef __attribute__((ext_vector_type(8)))  float  v8f;

// ---------------- helpers ----------------

__device__ __forceinline__ float block_reduce_sum(float v) {
    __shared__ float sh[8];
    const int lane = threadIdx.x & 31;
    const int w    = threadIdx.x >> 5;
    #pragma unroll
    for (int d = 16; d >= 1; d >>= 1) v += __shfl_down(v, d, 32);
    if (lane == 0) sh[w] = v;
    __syncthreads();
    const int nw = (blockDim.x + 31) >> 5;
    v = (threadIdx.x < nw) ? sh[threadIdx.x] : 0.0f;
    if (w == 0) {
        #pragma unroll
        for (int d = 4; d >= 1; d >>= 1) v += __shfl_down(v, d, 32);
    }
    return v; // valid on thread 0
}

// Async-copy one 16-code bf16 chunk (16 KB) from global into LDS.
// 128 threads x 8 iterations x 16 B per lane = 16 KB; tracked by ASYNCcnt.
__device__ __forceinline__ void issue_chunk_async(const bf16_t* __restrict__ ebf,
                                                  int cb, bf16_t* buf, int tid) {
    unsigned ldsbase = (unsigned)(uintptr_t)buf;   // shared aperture: low 32 bits = LDS addr
    const char* gbase = (const char*)(ebf + (size_t)cb * DD);
    #pragma unroll
    for (int q = 0; q < 8; ++q) {
        int u = q * 128 + tid;  // 16-byte unit within the 16 KB chunk
        unsigned long long ga = (unsigned long long)(uintptr_t)(gbase + u * 16);
        unsigned la = ldsbase + u * 16;
        asm volatile("global_load_async_to_lds_b128 %0, %1, off"
                     :: "v"(la), "v"(ga) : "memory");
    }
}

// ---------------- kernels ----------------

__global__ void k_zero(float4* p, int n4) {
    int g = blockIdx.x * blockDim.x + threadIdx.x;
    if (g < n4) { float4 z; z.x = z.y = z.z = z.w = 0.0f; p[g] = z; }
}

// One block per code row: bf16 conversion + squared norm.
__global__ __launch_bounds__(128) void k_embed(const float* __restrict__ emb,
                                               bf16_t* __restrict__ ebf,
                                               float* __restrict__ e2) {
    const int m = blockIdx.x, t = threadIdx.x;
    float4 v = ((const float4*)(emb + (size_t)m * DD))[t];
    union { bf16_t b[4]; unsigned long long u; } pk;
    pk.b[0] = (bf16_t)v.x; pk.b[1] = (bf16_t)v.y;
    pk.b[2] = (bf16_t)v.z; pk.b[3] = (bf16_t)v.w;
    *(unsigned long long*)(ebf + (size_t)m * DD + t * 4) = pk.u;
    float ss = v.x * v.x + v.y * v.y + v.z * v.z + v.w * v.w;
    ss = block_reduce_sum(ss);
    if (t == 0) e2[m] = ss;
}

// Distance + argmin via bf16 WMMA. 4 waves/block, 64 x-rows/block.
// B chunks staged block-wide into LDS with async-to-LDS copies (double-buffered).
__global__ __launch_bounds__(128) void k_dist(const float* __restrict__ x,
                                              const bf16_t* __restrict__ ebf,
                                              const float* __restrict__ e2,
                                              int* __restrict__ idx,
                                              float* __restrict__ idxf) {
    __shared__ bf16_t xs[64 * DD];        // 64 KB: x tile
    __shared__ bf16_t bs[2][16 * DD];     // 2 x 16 KB: double-buffered B chunks
    const int tid  = threadIdx.x;
    const int tile = blockIdx.x * 64;

    // Prologue: kick off async copy of B chunk 0.
    issue_chunk_async(ebf, 0, bs[0], tid);

    // Stage 64 rows of x into LDS as bf16 (overlaps with async B copy).
    const float4* xg = (const float4*)(x + (size_t)tile * DD);
    #pragma unroll 4
    for (int it = 0; it < 64; ++it) {
        int g = it * 128 + tid;            // float4 index (8192 total)
        float4 v = xg[g];
        union { bf16_t b[4]; unsigned long long u; } pk;
        pk.b[0] = (bf16_t)v.x; pk.b[1] = (bf16_t)v.y;
        pk.b[2] = (bf16_t)v.z; pk.b[3] = (bf16_t)v.w;
        *(unsigned long long*)(xs + g * 4) = pk.u;
    }
    __syncthreads();

    const int lane = tid & 31;
    const int wv   = tid >> 5;
    const int half = lane >> 4;   // 0 or 1
    const int l16  = lane & 15;
    const int rowbase = wv * 16;

    // A fragments for this wave's 16 rows, full K=512 (16 steps of K=32).
    // 16-bit A layout: lanes0-15 hold K {0..7,16..23}, lanes16-31 hold K {8..15,24..31}.
    v16bf a[16];
    const bf16_t* xrow = xs + (size_t)(rowbase + l16) * DD;
    #pragma unroll
    for (int s = 0; s < 16; ++s) {
        const bf16_t* p = xrow + 32 * s + 8 * half;
        v8bf lo = *(const v8bf*)p;
        v8bf hi = *(const v8bf*)(p + 16);
        a[s] = __builtin_shufflevector(lo, hi,
                 0,1,2,3,4,5,6,7,8,9,10,11,12,13,14,15);
    }

    float minv[8];
    int   mini[8];
    #pragma unroll
    for (int v = 0; v < 8; ++v) { minv[v] = 3.4e38f; mini[v] = 0; }

    // Sweep all 2048 codes in 16-wide chunks, B via LDS double buffer.
    for (int cb = 0; cb < MC; cb += 16) {
        const int p = (cb >> 4) & 1;
        const bool more = (cb + 16) < MC;
        if (more) issue_chunk_async(ebf, cb + 16, bs[p ^ 1], tid);
        // Current chunk's 8 asyncs complete when count drops to the 8 just issued.
        if (more) asm volatile("s_wait_asynccnt 0x8" ::: "memory");
        else      asm volatile("s_wait_asynccnt 0x0" ::: "memory");
        __syncthreads();

        const bf16_t* erow = bs[p] + (size_t)l16 * DD;
        v8f c0 = {}, c1 = {};
        #pragma unroll
        for (int s = 0; s < 16; s += 2) {
            // B 32x16 layout: lanes0-15 K {0..7,16..23}, lanes16-31 K {8..15,24..31}, col = lane&15.
            const bf16_t* q0 = erow + 32 * s + 8 * half;
            const bf16_t* q1 = erow + 32 * (s + 1) + 8 * half;
            v16bf b0 = __builtin_shufflevector(*(const v8bf*)q0, *(const v8bf*)(q0 + 16),
                 0,1,2,3,4,5,6,7,8,9,10,11,12,13,14,15);
            v16bf b1 = __builtin_shufflevector(*(const v8bf*)q1, *(const v8bf*)(q1 + 16),
                 0,1,2,3,4,5,6,7,8,9,10,11,12,13,14,15);
            c0 = __builtin_amdgcn_wmma_f32_16x16x32_bf16(
                     false, a[s], false, b0, (short)0, c0, false, false);
            c1 = __builtin_amdgcn_wmma_f32_16x16x32_bf16(
                     false, a[s + 1], false, b1, (short)0, c1, false, false);
        }
        float e2v = e2[cb + l16];
        #pragma unroll
        for (int v = 0; v < 8; ++v) {
            float acc = c0[v] + c1[v];
            float sc = e2v - 2.0f * acc;   // ||x||^2 constant per row: drop it
            if (sc < minv[v]) { minv[v] = sc; mini[v] = cb + l16; }
        }
        __syncthreads();   // before buffer p is refilled in the next iteration
    }

    // Reduce argmin across the 16 lanes of each half (xor 8/4/2/1 stays in-half).
    #pragma unroll
    for (int v = 0; v < 8; ++v) {
        float mv = minv[v]; int mi = mini[v];
        #pragma unroll
        for (int d = 8; d >= 1; d >>= 1) {
            float ov = __shfl_xor(mv, d, 32);
            int   oi = __shfl_xor(mi, d, 32);
            if (ov < mv || (ov == mv && oi < mi)) { mv = ov; mi = oi; }
        }
        if (l16 == 0) {
            int grow = tile + rowbase + v + 8 * half;   // C layout: VGPR v -> row v / v+8
            idx[grow]  = mi;
            idxf[grow] = (float)mi;
        }
    }
}

// Scatter: counts[idx] += 1 ; dw[idx] += x. One block per x row.
__global__ __launch_bounds__(128) void k_scatter(const float* __restrict__ x,
                                                 const int* __restrict__ idx,
                                                 float* __restrict__ counts,
                                                 float* __restrict__ dw) {
    const int n = blockIdx.x, t = threadIdx.x;
    const int m = idx[n];
    float4 v = ((const float4*)(x + (size_t)n * DD))[t];
    float* p = dw + (size_t)m * DD + t * 4;
    atomicAdd(p + 0, v.x);
    atomicAdd(p + 1, v.y);
    atomicAdd(p + 2, v.z);
    atomicAdd(p + 3, v.w);
    if (t == 0) atomicAdd(&counts[m], 1.0f);
}

// quant_st = embedding[idx] ; accumulate sum((x - q)^2).
__global__ __launch_bounds__(256) void k_quant(const float* __restrict__ x,
                                               const float* __restrict__ emb,
                                               const int* __restrict__ idx,
                                               float* __restrict__ qout,
                                               float* __restrict__ scal) {
    const int g = blockIdx.x * 256 + threadIdx.x;   // float4 index, 4194304 total
    const int n = g >> 7;
    const int c = g & 127;
    const int m = idx[n];
    float4 q  = ((const float4*)emb)[(size_t)m * 128 + c];
    float4 xv = ((const float4*)x)[g];
    ((float4*)qout)[g] = q;
    float dx = xv.x - q.x, dy = xv.y - q.y, dz = xv.z - q.z, dwv = xv.w - q.w;
    float ss = dx * dx + dy * dy + dz * dz + dwv * dwv;
    ss = block_reduce_sum(ss);
    if (threadIdx.x == 0) atomicAdd(&scal[0], ss);
}

__global__ __launch_bounds__(256) void k_count_ema(const float* __restrict__ ema_count,
                                                   const float* __restrict__ counts,
                                                   float* __restrict__ nc_pre,
                                                   float* __restrict__ scal) {
    const int m = blockIdx.x * 256 + threadIdx.x;
    float nc = DECAY_F * ema_count[m] + ONEM_F * counts[m];
    nc_pre[m] = nc;
    float s = block_reduce_sum(nc);
    if (threadIdx.x == 0) atomicAdd(&scal[1], s);
}

__global__ __launch_bounds__(256) void k_count_final(const float* __restrict__ nc_pre,
                                                     const float* __restrict__ scal,
                                                     float* __restrict__ ncnt_out,
                                                     float* __restrict__ nc_s) {
    const int m = blockIdx.x * 256 + threadIdx.x;
    const float n = scal[1];
    float v = (nc_pre[m] + EPS_F) / (n + (float)MC * EPS_F) * n;
    ncnt_out[m] = v;
    nc_s[m] = v;
}

__global__ __launch_bounds__(256) void k_weight(const float* __restrict__ ema_weight,
                                                const float* __restrict__ dw,
                                                const float* __restrict__ nc_s,
                                                float* __restrict__ nw_out,
                                                float* __restrict__ ne_out) {
    const int g = blockIdx.x * 256 + threadIdx.x;  // float4 index, 262144 total
    const int m = g >> 7;
    float4 w = ((const float4*)ema_weight)[g];
    float4 d = ((const float4*)dw)[g];
    float4 nw;
    nw.x = DECAY_F * w.x + ONEM_F * d.x;
    nw.y = DECAY_F * w.y + ONEM_F * d.y;
    nw.z = DECAY_F * w.z + ONEM_F * d.z;
    nw.w = DECAY_F * w.w + ONEM_F * d.w;
    ((float4*)nw_out)[g] = nw;
    float inv = 1.0f / nc_s[m];
    float4 ne; ne.x = nw.x * inv; ne.y = nw.y * inv; ne.z = nw.z * inv; ne.w = nw.w * inv;
    ((float4*)ne_out)[g] = ne;
}

// Stable select: used codes (ascending) first, then unused; first 128.
__global__ void k_select(const float* __restrict__ counts,
                         int* __restrict__ sel, float* __restrict__ valid,
                         float* __restrict__ scal) {
    int cnt = 0;
    for (int m = 0; m < MC && cnt < 128; ++m)
        if (counts[m] > 0.0f) { sel[cnt] = m; valid[cnt] = 1.0f; ++cnt; }
    const int nv = cnt;
    for (int m = 0; m < MC && cnt < 128; ++m)
        if (!(counts[m] > 0.0f)) { sel[cnt] = m; valid[cnt] = 0.0f; ++cnt; }
    scal[3] = (float)nv;
}

__global__ __launch_bounds__(128) void k_normed(const float* __restrict__ ne_out,
                                                const int* __restrict__ sel,
                                                const float* __restrict__ valid,
                                                float* __restrict__ normed) {
    const int i = blockIdx.x, t = threadIdx.x;
    const int m = sel[i];
    const float vld = valid[i];
    float4 v = ((const float4*)ne_out)[(size_t)m * 128 + t];
    float ss = v.x * v.x + v.y * v.y + v.z * v.z + v.w * v.w;
    ss = block_reduce_sum(ss);
    __shared__ float nrm;
    if (t == 0) nrm = fmaxf(sqrtf(ss), 1e-12f);
    __syncthreads();
    float sc = vld / nrm;
    float4 o; o.x = v.x * sc; o.y = v.y * sc; o.z = v.z * sc; o.w = v.w * sc;
    ((float4*)normed)[i * 128 + t] = o;
}

__global__ __launch_bounds__(128) void k_cosine(const float* __restrict__ normed,
                                                const float* __restrict__ valid,
                                                float* __restrict__ scal) {
    __shared__ float ni[DD];
    const int i = blockIdx.x, t = threadIdx.x;
    float4 v = ((const float4*)normed)[i * 128 + t];
    ((float4*)ni)[t] = v;
    __syncthreads();
    const int j = t;
    const float4* nj = (const float4*)(normed + (size_t)j * DD);
    float dot = 0.0f;
    #pragma unroll 8
    for (int k = 0; k < 128; ++k) {
        float4 a = ((const float4*)ni)[k];
        float4 b = nj[k];
        dot += a.x * b.x + a.y * b.y + a.z * b.z + a.w * b.w;
    }
    float dterm = (j == i) ? valid[i] : 0.0f;
    float d = dot - dterm;
    float sq = d * d;
    sq = block_reduce_sum(sq);
    if (t == 0) atomicAdd(&scal[2], sq);
}

__global__ void k_losses(const float* __restrict__ scal, float* __restrict__ loss_out) {
    float mse = scal[0] / (float)ND_TOT;
    loss_out[0] = COMMIT_F * mse;     // commitment_loss
    loss_out[1] = mse;                // codebook_loss
    float nv = scal[3];
    loss_out[2] = scal[2] / (nv * nv) * ORTHO_WF;   // ortho_loss
}

// ---------------- launch ----------------

extern "C" void kernel_launch(void* const* d_in, const int* in_sizes, int n_in,
                              void* d_out, int out_size, void* d_ws, size_t ws_size,
                              hipStream_t stream) {
    const float* x          = (const float*)d_in[0];   // (32,1024,512)
    const float* embedding  = (const float*)d_in[1];   // (2048,512)
    const float* ema_count  = (const float*)d_in[2];   // (2048,)
    const float* ema_weight = (const float*)d_in[3];   // (2048,512)

    char* ws = (char*)d_ws;
    bf16_t* e_bf   = (bf16_t*)(ws + 0);          // 2,097,152 B
    float*  e2     = (float*)(ws + 2097152);     // 8,192 B
    int*    idx    = (int*)  (ws + 2105344);     // 131,072 B
    float*  counts = (float*)(ws + 2236416);     // 8,192 B
    float*  dw     = (float*)(ws + 2244608);     // 4,194,304 B
    float*  nc_pre = (float*)(ws + 6438912);     // 8,192 B
    float*  nc_s   = (float*)(ws + 6447104);     // 8,192 B
    float*  scal   = (float*)(ws + 6455296);     // 64 B  [mse, nsum, ortho, nvalid]
    int*    sel    = (int*)  (ws + 6455360);     // 512 B
    float*  valid  = (float*)(ws + 6455872);     // 512 B
    float*  normed = (float*)(ws + 6456384);     // 262,144 B

    float* out       = (float*)d_out;
    float* q_out     = out;                  // 16,777,216
    float* loss_out  = out + 16777216;       // 3
    float* idxf_out  = out + 16777219;       // 32,768
    float* ne_out    = out + 16809987;       // 1,048,576
    float* ncnt_out  = out + 17858563;       // 2,048
    float* nw_out    = out + 17860611;       // 1,048,576

    // 1. zero counts/dw/nc/scalars region: bytes [2236416, 6455424) = 4,219,008 B
    {
        int n4 = 4219008 / 16;   // 263,688 float4
        k_zero<<<(n4 + 255) / 256, 256, 0, stream>>>((float4*)(ws + 2236416), n4);
    }
    // 2. embedding -> bf16 + norms
    k_embed<<<MC, 128, 0, stream>>>(embedding, e_bf, e2);
    // 3. WMMA distances + argmin (async-to-LDS B staging)
    k_dist<<<NN / 64, 128, 0, stream>>>(x, e_bf, e2, idx, idxf_out);
    // 4. scatter counts + dw
    k_scatter<<<NN, 128, 0, stream>>>(x, idx, counts, dw);
    // 5. quantized gather + MSE
    k_quant<<<(ND_TOT / 4) / 256, 256, 0, stream>>>(x, embedding, idx, q_out, scal);
    // 6. EMA counts + sum
    k_count_ema<<<MC / 256, 256, 0, stream>>>(ema_count, counts, nc_pre, scal);
    // 7. Laplace smoothing
    k_count_final<<<MC / 256, 256, 0, stream>>>(nc_pre, scal, ncnt_out, nc_s);
    // 8. EMA weights + new embedding
    k_weight<<<(MC * DD / 4) / 256, 256, 0, stream>>>(ema_weight, dw, nc_s, nw_out, ne_out);
    // 9. select first 128 (used-first, stable)
    k_select<<<1, 1, 0, stream>>>(counts, sel, valid, scal);
    // 10. normalize selected rows
    k_normed<<<128, 128, 0, stream>>>(ne_out, sel, valid, normed);
    // 11. cosine-sim + ortho accumulation
    k_cosine<<<128, 128, 0, stream>>>(normed, valid, scal);
    // 12. final scalars
    k_losses<<<1, 1, 0, stream>>>(scal, loss_out);
}
